// KblamLlamaAttention_43233140802081
// MI455X (gfx1250) — matrix operand order repaired and checked by reading, not compiled, not run on hardware
//
#include <hip/hip_runtime.h>
#include <hip/hip_bf16.h>
#include <math.h>

#define SEQ   1024
#define HID   4096
#define NH    32
#define HD    128
#define KBN   2048
#define KBROW 8192          // NUM_SLOTS * HID
#define TOPKN 100
#define TKPAD 128
#define SCALE 0.08838834764831845f
#define NEGBIG -1e30f

typedef __attribute__((ext_vector_type(16))) __bf16 v16bf;
typedef __attribute__((ext_vector_type(8)))  __bf16 v8bf;
typedef __attribute__((ext_vector_type(8)))  float  v8f;
typedef __attribute__((ext_vector_type(4)))  unsigned int u32x4;
typedef __attribute__((ext_vector_type(8)))  int    i32x8;
typedef __attribute__((ext_vector_type(4)))  int    i32x4;

__device__ __forceinline__ v8f v8f_zero() {
  v8f z;
#pragma unroll
  for (int i = 0; i < 8; i++) z[i] = 0.f;
  return z;
}

__device__ __forceinline__ v8f wmma_bf16f32(v16bf a, v16bf b, v8f c) {
  return __builtin_amdgcn_wmma_f32_16x16x32_bf16(false, a, false, b, (short)0, c,
                                                 false, false);
}

// Build a 16x32 bf16 A-fragment for this lane.
// rowbase points at (row = lane%16, k = chunk base). Contiguous k order, same as the B loads:
// lanes 0-15 hold K = 0..15 ; lanes 16-31 hold K = 16..31.
__device__ __forceinline__ v16bf make_afrag(const __bf16* rowbase, int lhalf) {
  const v8bf lo = *(const v8bf*)(rowbase + lhalf * 16);
  const v8bf hi = *(const v8bf*)(rowbase + lhalf * 16 + 8);
  v16bf a;
#pragma unroll
  for (int i = 0; i < 8; i++) { a[i] = lo[i]; a[i + 8] = hi[i]; }
  return a;
}

// ---- Tensor Data Mover: 2D bf16 tile -> LDS (6-arg builtin, clang-23) ----
// D# built per CDNA5 ISA §8: group0 (flags/lds/global addr), group1 (dims).
__device__ __forceinline__ void tdm_load_2d(unsigned lds_off, const void* gptr,
                                            unsigned tile_d0, unsigned tile_d1,
                                            unsigned stride_d0,
                                            unsigned tensor_d0, unsigned tensor_d1) {
  unsigned long long ga = (unsigned long long)gptr;
  u32x4 g0;
  g0[0] = 1u;                                   // count=1 (valid), no gather
  g0[1] = lds_off;                              // lds_addr
  g0[2] = (unsigned)(ga & 0xffffffffu);         // global_addr[31:0]
  g0[3] = (unsigned)((ga >> 32) & 0x1ffffffu) | (2u << 30);  // addr hi | type=2
  i32x8 g1;
  g1[0] = (int)(1u << 16);                      // workgroup_mask=0, data_size=1 (2B)
  g1[1] = (int)((tensor_d0 & 0xffffu) << 16);   // tensor_dim0[15:0] @ bits 63:48
  g1[2] = (int)(((tensor_d0 >> 16) & 0xffffu) | ((tensor_d1 & 0xffffu) << 16));
  g1[3] = (int)(((tensor_d1 >> 16) & 0xffffu) | ((tile_d0 & 0xffffu) << 16));
  g1[4] = (int)(tile_d1 & 0xffffu);             // tile_dim1 | tile_dim2=0
  g1[5] = (int)stride_d0;                       // tensor_dim0_stride[31:0]
  g1[6] = 0;
  g1[7] = 0;
  i32x4 z4; z4[0] = 0; z4[1] = 0; z4[2] = 0; z4[3] = 0;
  i32x8 z8;
#pragma unroll
  for (int i = 0; i < 8; i++) z8[i] = 0;
  __builtin_amdgcn_tensor_load_to_lds(g0, g1, z4, z4, z8, 0);
}

// ---------------------------------------------------------------- cvt fp32->bf16
__global__ void cvt_kernel(const float* __restrict__ src, __bf16* __restrict__ dst,
                           int n) {
  int i = blockIdx.x * 256 + threadIdx.x;
  if (i < n) dst[i] = (__bf16)src[i];
}

// ---------------------------------------------------------------- GEMM (WMMA)
// C[M,N] f32 = A[M,K] bf16 (row major) * B[N,K] bf16 (row major, i.e. X @ W^T)
__global__ __launch_bounds__(256) void gemm_bf16_kernel(
    const __bf16* __restrict__ A, const __bf16* __restrict__ B,
    float* __restrict__ C, int M, int N, int K) {
  __shared__ __attribute__((aligned(32))) __bf16 ldsA[128 * 32];
  __shared__ __attribute__((aligned(32))) __bf16 ldsB[128 * 32];
  const int tid = threadIdx.x;
  const int lane = tid & 31, w = tid >> 5;
  const int lcol = lane & 15, lhalf = lane >> 4;
  const int m0 = blockIdx.x * 128, n0 = blockIdx.y * 128;

  v8f acc[8];
#pragma unroll
  for (int nt = 0; nt < 8; nt++) acc[nt] = v8f_zero();

  const int srow = tid >> 1;                 // 0..127
  const int h16  = (tid & 1) * 16;           // 0 or 16

  for (int k0 = 0; k0 < K; k0 += 32) {
    __syncthreads();
    const uint4* sa = (const uint4*)(A + (size_t)(m0 + srow) * K + k0 + h16);
    uint4* da = (uint4*)&ldsA[srow * 32 + h16];
    da[0] = sa[0]; da[1] = sa[1];
    const uint4* sb = (const uint4*)(B + (size_t)(n0 + srow) * K + k0 + h16);
    uint4* db = (uint4*)&ldsB[srow * 32 + h16];
    db[0] = sb[0]; db[1] = sb[1];
    __syncthreads();

    v16bf a = make_afrag(&ldsA[(w * 16 + lcol) * 32], lhalf);
#pragma unroll
    for (int nt = 0; nt < 8; nt++) {
      v16bf b = *(const v16bf*)&ldsB[(nt * 16 + lcol) * 32 + lhalf * 16];
      acc[nt] = wmma_bf16f32(a, b, acc[nt]);
    }
  }

#pragma unroll
  for (int nt = 0; nt < 8; nt++)
#pragma unroll
    for (int r = 0; r < 8; r++) {
      int row = m0 + w * 16 + r + 8 * lhalf;
      int col = n0 + nt * 16 + lcol;
      C[(size_t)row * N + col] = acc[nt][r];
    }
}

// ------------------------------------------------- RoPE + pack [S,HID]->[H,S,D] bf16
__global__ void rope_pack_kernel(const float* __restrict__ x,
                                 const int* __restrict__ pos,
                                 __bf16* __restrict__ out, int doRope) {
  int idx = blockIdx.x * 256 + threadIdx.x;
  if (idx >= SEQ * HID) return;
  int s = idx / HID, rem = idx % HID;
  int hh = rem / HD, d = rem % HD;
  float v = x[idx];
  if (doRope) {
    int j = d & 63;
    float inv = __powf(10000.f, -(float)j * (1.f / 64.f));
    float f = (float)pos[s] * inv;
    float partner = (d < 64) ? -x[(size_t)s * HID + rem + 64]
                             :  x[(size_t)s * HID + rem - 64];
    v = v * __cosf(f) + partner * __sinf(f);
  }
  out[((size_t)hh * SEQ + s) * HD + d] = (__bf16)v;
}

// ------------------------------------------------- column sum of q2 (for KB score)
__global__ void colsum_kernel(const float* __restrict__ x, float* __restrict__ out) {
  int col = blockIdx.x * 256 + threadIdx.x;
  float acc = 0.f;
  for (int s = 0; s < SEQ; s++) acc += x[(size_t)s * HID + col];
  out[col] = acc;
}

// ------------------------------------------------- kb score: dot(colsum, kb row)
__global__ void kbscore_kernel(const float* __restrict__ sq,
                               const float* __restrict__ kb,
                               float* __restrict__ sc) {
  int lane = threadIdx.x & 31;
  int e = blockIdx.x * 8 + (threadIdx.x >> 5);
  const float* row = kb + (size_t)e * KBROW;   // slot 0 = first HID
  float acc = 0.f;
  for (int i = lane; i < HID; i += 32) acc += sq[i] * row[i];
#pragma unroll
  for (int off = 16; off; off >>= 1) acc += __shfl_xor(acc, off, 32);
  if (lane == 0) sc[e] = acc;
}

// ------------------------------------------------- rank-based deterministic top-k
__global__ void topk_kernel(const float* __restrict__ sc, int* __restrict__ topidx) {
  __shared__ float s[KBN];
  for (int i = threadIdx.x; i < KBN; i += 256) s[i] = sc[i];
  __syncthreads();
  for (int e = threadIdx.x; e < KBN; e += 256) {
    float v = s[e];
    int rank = 0;
    for (int j = 0; j < KBN; j++) {
      float u = s[j];
      rank += (u > v) || (u == v && j < e);
    }
    if (rank < TOPKN) topidx[rank] = e;   // unique ranks -> no atomics
  }
}

// ------------------------------------------------- gather top-k KB rows -> bf16 [H,128,D]
__global__ void gather_kernel(const float* __restrict__ kb,
                              const int* __restrict__ topidx,
                              __bf16* __restrict__ out) {
  int idx = blockIdx.x * 256 + threadIdx.x;
  if (idx >= NH * TKPAD * HD) return;
  int hh = idx / (TKPAD * HD);
  int t = (idx / HD) % TKPAD;
  int d = idx % HD;
  float v = 0.f;
  if (t < TOPKN) v = kb[(size_t)topidx[t] * KBROW + hh * HD + d];
  out[idx] = (__bf16)v;
}

// ------------------------------------------------- flash attention (WMMA + TDM)
__global__ __launch_bounds__(256) void attn_kernel(
    const __bf16* __restrict__ Qh, const __bf16* __restrict__ Q2h,
    const __bf16* __restrict__ Kh, const __bf16* __restrict__ Vh,
    const __bf16* __restrict__ kbK, const __bf16* __restrict__ kbV,
    const float* __restrict__ shift, __bf16* __restrict__ attnO) {
  __shared__ __attribute__((aligned(32))) __bf16 ldsK[32 * 128];    // [key][d]
  __shared__ __attribute__((aligned(32))) __bf16 ldsVt[128 * 32];   // [d][key]
  __shared__ __attribute__((aligned(32))) __bf16 ldsP[8][16 * 32];  // per-wave P

  const int tid = threadIdx.x;
  const int lane = tid & 31, w = tid >> 5;
  const int lcol = lane & 15, lhalf = lane >> 4;
  const int h = blockIdx.y;
  const int s0blk = blockIdx.x * 128;
  const int sW = s0blk + w * 16;          // first query row of this wave
  const int rowMine = sW + lhalf * 8;     // + r -> absolute query row

  float mrow[8], lrow[8];
  v8f o[8];
#pragma unroll
  for (int r = 0; r < 8; r++) { mrow[r] = -3e38f; lrow[r] = 0.f; }
#pragma unroll
  for (int nt = 0; nt < 8; nt++) o[nt] = v8f_zero();

  v16bf aQ[4];

  auto stage = [&](const __bf16* ksrc, const __bf16* vsrc) {
    __syncthreads();                       // protect LDS reuse
    if (tid < 32) {                        // one TDM issue per block (wave 0)
      tdm_load_2d((unsigned)(unsigned long long)(void*)ldsK, ksrc,
                  /*tile*/ 128, 32, /*stride*/ 128, /*tensor*/ 128, 32);
    }
    {  // transpose V tile into ldsVt: thread -> (key, 16-dim chunk)
      int key = tid & 31, dc = tid >> 5;
      const __bf16* sp = vsrc + key * 128 + dc * 16;
#pragma unroll
      for (int i = 0; i < 16; i++) ldsVt[(dc * 16 + i) * 32 + key] = sp[i];
    }
    if (tid < 32) __builtin_amdgcn_s_wait_tensorcnt(0);
    __syncthreads();
  };

  auto process = [&](float bias, int colbase, bool kbMask, bool causal) {
    // S tile: 16 rows x 32 keys = two C fragments
    v8f s0 = v8f_zero(), s1 = v8f_zero();
#pragma unroll
    for (int c = 0; c < 4; c++) {
      v16bf b0 = *(const v16bf*)&ldsK[lcol * 128 + c * 32 + lhalf * 16];
      v16bf b1 = *(const v16bf*)&ldsK[(lcol + 16) * 128 + c * 32 + lhalf * 16];
      s0 = wmma_bf16f32(aQ[c], b0, s0);
      s1 = wmma_bf16f32(aQ[c], b1, s1);
    }
    const int col0 = colbase + lcol, col1 = colbase + 16 + lcol;
    float p0[8], p1[8];
#pragma unroll
    for (int r = 0; r < 8; r++) {
      float x0 = s0[r] * SCALE + bias;
      float x1 = s1[r] * SCALE + bias;
      if (kbMask) {
        if (col0 >= TOPKN) x0 = NEGBIG;
        if (col1 >= TOPKN) x1 = NEGBIG;
      }
      if (causal) {
        int row = rowMine + r;
        if (col0 > row) x0 = NEGBIG;
        if (col1 > row) x1 = NEGBIG;
      }
      p0[r] = x0; p1[r] = x1;
    }
    // streaming softmax (row reductions across the 16 lanes of each half)
#pragma unroll
    for (int r = 0; r < 8; r++) {
      float tmax = fmaxf(p0[r], p1[r]);
#pragma unroll
      for (int off = 1; off < 16; off <<= 1)
        tmax = fmaxf(tmax, __shfl_xor(tmax, off, 32));
      float nm = fmaxf(mrow[r], tmax);
      float corr = __expf(mrow[r] - nm);
      float e0 = __expf(p0[r] - nm), e1 = __expf(p1[r] - nm);
      float rs = e0 + e1;
#pragma unroll
      for (int off = 1; off < 16; off <<= 1) rs += __shfl_xor(rs, off, 32);
      lrow[r] = lrow[r] * corr + rs;
      mrow[r] = nm;
      p0[r] = e0; p1[r] = e1;
#pragma unroll
      for (int nt = 0; nt < 8; nt++) o[nt][r] *= corr;
    }
    // P (C-layout) -> per-wave LDS scratch -> A-fragment layout
    __bf16* ps = &ldsP[w][0];
#pragma unroll
    for (int r = 0; r < 8; r++) {
      int row = r + 8 * lhalf;
      ps[row * 32 + lcol]      = (__bf16)p0[r];
      ps[row * 32 + 16 + lcol] = (__bf16)p1[r];
    }
    v16bf aP = make_afrag(ps + lcol * 32, lhalf);
#pragma unroll
    for (int nt = 0; nt < 8; nt++) {
      v16bf bV = *(const v16bf*)&ldsVt[(nt * 16 + lcol) * 32 + lhalf * 16];
      o[nt] = wmma_bf16f32(aP, bV, o[nt]);
    }
  };

  // ---- KB phase: q2 logits + score_shift, columns >= TOPK masked ----
#pragma unroll
  for (int c = 0; c < 4; c++)
    aQ[c] = make_afrag(Q2h + ((size_t)h * SEQ + (sW + lcol)) * HD + c * 32, lhalf);
  const float bias = shift[h];
  for (int j = 0; j < 4; j++) {
    stage(kbK + ((size_t)h * TKPAD + j * 32) * HD,
          kbV + ((size_t)h * TKPAD + j * 32) * HD);
    process(bias, j * 32, true, false);
  }

  // ---- causal phase: q logits, causal mask ----
#pragma unroll
  for (int c = 0; c < 4; c++)
    aQ[c] = make_afrag(Qh + ((size_t)h * SEQ + (sW + lcol)) * HD + c * 32, lhalf);
  const int jmax = blockIdx.x * 4 + 4;
  for (int j = 0; j < jmax; j++) {
    stage(Kh + ((size_t)h * SEQ + j * 32) * HD,
          Vh + ((size_t)h * SEQ + j * 32) * HD);
    process(0.f, j * 32, false, true);
  }

  // ---- epilogue: normalize, write bf16 [S, H*D] for final GEMM ----
#pragma unroll
  for (int nt = 0; nt < 8; nt++)
#pragma unroll
    for (int r = 0; r < 8; r++) {
      int row = sW + r + 8 * lhalf;
      float val = o[nt][r] / lrow[r];
      attnO[(size_t)row * HID + h * HD + nt * 16 + lcol] = (__bf16)val;
    }
}

// =========================================================================
extern "C" void kernel_launch(void* const* d_in, const int* in_sizes, int n_in,
                              void* d_out, int out_size, void* d_ws, size_t ws_size,
                              hipStream_t stream) {
  const float* hidden = (const float*)d_in[0];
  // d_in[1] = attention_mask (causal handled analytically)
  const int* pos = (const int*)d_in[2];
  const float* kbk = (const float*)d_in[3];
  const float* kbv = (const float*)d_in[4];
  const float* Wq = (const float*)d_in[5];
  const float* Wq2 = (const float*)d_in[6];
  const float* Wk = (const float*)d_in[7];
  const float* Wv = (const float*)d_in[8];
  const float* Wo = (const float*)d_in[9];
  const float* shift = (const float*)d_in[10];
  float* out = (float*)d_out;

  char* ws = (char*)d_ws;
  size_t off = 0;
  auto alloc = [&](size_t bytes) -> void* {
    off = (off + 255) & ~(size_t)255;
    void* p = ws + off;
    off += bytes;
    return p;
  };
  const size_t WEL = (size_t)HID * HID;       // 16M
  const size_t XEL = (size_t)SEQ * HID;       // 4M
  __bf16* Wqb  = (__bf16*)alloc(WEL * 2);
  __bf16* Wq2b = (__bf16*)alloc(WEL * 2);
  __bf16* Wkb  = (__bf16*)alloc(WEL * 2);
  __bf16* Wvb  = (__bf16*)alloc(WEL * 2);
  __bf16* Wob  = (__bf16*)alloc(WEL * 2);
  __bf16* Xb   = (__bf16*)alloc(XEL * 2);
  float* qf  = (float*)alloc(XEL * 4);
  float* q2f = (float*)alloc(XEL * 4);
  float* kf  = (float*)alloc(XEL * 4);
  float* vf  = (float*)alloc(XEL * 4);
  __bf16* QhB  = (__bf16*)alloc(XEL * 2);
  __bf16* Q2hB = (__bf16*)alloc(XEL * 2);
  __bf16* KhB  = (__bf16*)alloc(XEL * 2);
  __bf16* VhB  = (__bf16*)alloc(XEL * 2);
  __bf16* kbKb = (__bf16*)alloc((size_t)NH * TKPAD * HD * 2);
  __bf16* kbVb = (__bf16*)alloc((size_t)NH * TKPAD * HD * 2);
  __bf16* attnOb = (__bf16*)alloc(XEL * 2);
  float* sumq2 = (float*)alloc(HID * 4);
  float* scores = (float*)alloc(KBN * 4);
  int* topidx = (int*)alloc(TKPAD * 4);

  // 1. fp32 -> bf16 conversions
  cvt_kernel<<<(int)((XEL + 255) / 256), 256, 0, stream>>>(hidden, Xb, (int)XEL);
  cvt_kernel<<<(int)((WEL + 255) / 256), 256, 0, stream>>>(Wq, Wqb, (int)WEL);
  cvt_kernel<<<(int)((WEL + 255) / 256), 256, 0, stream>>>(Wq2, Wq2b, (int)WEL);
  cvt_kernel<<<(int)((WEL + 255) / 256), 256, 0, stream>>>(Wk, Wkb, (int)WEL);
  cvt_kernel<<<(int)((WEL + 255) / 256), 256, 0, stream>>>(Wv, Wvb, (int)WEL);
  cvt_kernel<<<(int)((WEL + 255) / 256), 256, 0, stream>>>(Wo, Wob, (int)WEL);

  // 2. projections (X @ W^T) via WMMA
  dim3 gg(SEQ / 128, HID / 128);
  gemm_bf16_kernel<<<gg, 256, 0, stream>>>(Xb, Wqb, qf, SEQ, HID, HID);
  gemm_bf16_kernel<<<gg, 256, 0, stream>>>(Xb, Wq2b, q2f, SEQ, HID, HID);
  gemm_bf16_kernel<<<gg, 256, 0, stream>>>(Xb, Wkb, kf, SEQ, HID, HID);
  gemm_bf16_kernel<<<gg, 256, 0, stream>>>(Xb, Wvb, vf, SEQ, HID, HID);

  // 3. RoPE (q,k) + head-major bf16 pack (all four)
  int gp = (SEQ * HID) / 256;
  rope_pack_kernel<<<gp, 256, 0, stream>>>(qf, pos, QhB, 1);
  rope_pack_kernel<<<gp, 256, 0, stream>>>(kf, pos, KhB, 1);
  rope_pack_kernel<<<gp, 256, 0, stream>>>(q2f, pos, Q2hB, 0);
  rope_pack_kernel<<<gp, 256, 0, stream>>>(vf, pos, VhB, 0);

  // 4-6. KB scoring (factorized) + deterministic top-k
  colsum_kernel<<<HID / 256, 256, 0, stream>>>(q2f, sumq2);
  kbscore_kernel<<<KBN / 8, 256, 0, stream>>>(sumq2, kbk, scores);
  topk_kernel<<<1, 256, 0, stream>>>(scores, topidx);

  // 7. gather top-k KB keys/values (bf16, padded to 128)
  int gg2 = (NH * TKPAD * HD) / 256;
  gather_kernel<<<gg2, 256, 0, stream>>>(kbk, topidx, kbKb);
  gather_kernel<<<gg2, 256, 0, stream>>>(kbv, topidx, kbVb);

  // 8. flash attention over [KB(100 padded 128) | causal 1024]
  dim3 ga(SEQ / 128, NH);
  attn_kernel<<<ga, 256, 0, stream>>>(QhB, Q2hB, KhB, VhB, kbKb, kbVb, shift,
                                      attnOb);

  // 9. output projection -> d_out (f32)
  gemm_bf16_kernel<<<gg, 256, 0, stream>>>(attnOb, Wob, out, SEQ, HID, HID);
  (void)in_sizes; (void)n_in; (void)out_size; (void)ws_size;
}
